// Atten_84928683311946
// MI455X (gfx1250) — compile-verified, matrix operand order checked by reading, not run on hardware
//
#include <hip/hip_runtime.h>
#include <hip/hip_bf16.h>

typedef __bf16 v16bf __attribute__((ext_vector_type(16)));
typedef float  v8f   __attribute__((ext_vector_type(8)));
typedef __bf16 bf4   __attribute__((ext_vector_type(4)));

#define C_DIM 512
#define N_TOK 256
#define HEADS 8
#define DH    64
#define BB    4
#define NN    8192
#define SPLIT 8
#define CHUNK (NN / SPLIT)                    // 1024 rows per split
#define PART_STRIDE (N_TOK * DH + 2 * N_TOK)  // 16896 floats per partial

// ---------------------------------------------------------------- WMMA helpers
static __device__ __forceinline__ v8f wmma_bf16(v16bf a, v16bf b, v8f c) {
  return __builtin_amdgcn_wmma_f32_16x16x32_bf16(false, a, false, b, (short)0, c,
                                                 false, false);
}

// A fragment: 16x32, row-major source [row][k], stride ld (per lane k = 8h+j and 16+8h+j, table order).
template <typename T>
static __device__ __forceinline__ v16bf frag_A(const T* p, int ld, int r0, int k0,
                                               int lane) {
  const T* s = p + (size_t)(r0 + (lane & 15)) * ld + (k0 + ((lane >> 4) << 3));
  v16bf f;
#pragma unroll
  for (int j = 0; j < 8; ++j) {
    f[j]     = (__bf16)s[j];
    f[j + 8] = (__bf16)s[j + 16];
  }
  return f;
}

// Swizzled B layout: [tile][lane 0..31][16 contiguous bf16], tile = 512 elems.
static __device__ __forceinline__ int swz_tile(int kin, int nin) {
  return (nin + ((kin & 8) ? 16 : 0)) * 16 + (kin & 7) + ((kin >> 4) << 3);
}
static __device__ __forceinline__ v16bf frag_Bs(const __bf16* p, int tile,
                                                int lane) {
  return *(const v16bf*)(p + ((size_t)tile * 32 + lane) * 16);
}
static __device__ __forceinline__ void stB_swz(__bf16* p, int k, int n, __bf16 v) {
  p[(size_t)(((k >> 5) * 4 + (n >> 4)) * 512) + swz_tile(k & 31, n & 15)] = v;
}

// ----------------------- CDNA5 async global->LDS copy (ASYNCcnt, VGLOBAL GVS)
static __device__ __forceinline__ void async_cp16(unsigned lds_addr,
                                                  const void* gbase,
                                                  unsigned goff) {
  asm volatile("global_load_async_to_lds_b128 %0, %1, %2"
               :: "v"(lds_addr), "v"(goff), "s"(gbase)
               : "memory");
}
static __device__ __forceinline__ void async_wait0() {
  asm volatile("s_wait_asynccnt 0" ::: "memory");
}
static __device__ __forceinline__ unsigned lds_addr_of(const void* p) {
  return (unsigned)(size_t)p;  // LDS aperture: low 32 bits = LDS byte offset
}

// ----------------------------------------------------- kernel 0: weight prep
__global__ void k_cvt_lin(const float* __restrict__ src, __bf16* __restrict__ dst,
                          int n) {
  for (int i = blockIdx.x * blockDim.x + threadIdx.x; i < n;
       i += gridDim.x * blockDim.x)
    dst[i] = (__bf16)src[i];
}
// kv1_w (512x1024) -> per (mat,head) 64-tile swizzled regions of 32768 elems
__global__ void k_cvt_kv(const float* __restrict__ src, __bf16* __restrict__ dst) {
  const int n = C_DIM * 2 * C_DIM;
  for (int i = blockIdx.x * blockDim.x + threadIdx.x; i < n;
       i += gridDim.x * blockDim.x) {
    int k = i >> 10, col = i & 1023;
    int mat = col >> 9, h = (col >> 6) & 7, nt = (col >> 4) & 3, nin = col & 15;
    int kt = k >> 5, kin = k & 31;
    dst[(size_t)(((mat * HEADS + h) * 16 + kt) * 4 + nt) * 512 +
        swz_tile(kin, nin)] = (__bf16)src[i];
  }
}
// q3_w (512x512) -> per-head 64-tile swizzled regions of 32768 elems
__global__ void k_cvt_q3(const float* __restrict__ src, __bf16* __restrict__ dst) {
  const int n = C_DIM * C_DIM;
  for (int i = blockIdx.x * blockDim.x + threadIdx.x; i < n;
       i += gridDim.x * blockDim.x) {
    int k = i >> 9, col = i & 511;
    int h = col >> 6, nt = (col >> 4) & 3, nin = col & 15;
    int kt = k >> 5, kin = k & 31;
    dst[(size_t)((h * 16 + kt) * 4 + nt) * 512 + swz_tile(kin, nin)] =
        (__bf16)src[i];
  }
}
// out_w (512x512) -> per 128-col chunk, 128-tile swizzled regions of 65536 elems
__global__ void k_cvt_out(const float* __restrict__ src, __bf16* __restrict__ dst) {
  const int n = C_DIM * C_DIM;
  for (int i = blockIdx.x * blockDim.x + threadIdx.x; i < n;
       i += gridDim.x * blockDim.x) {
    int k = i >> 9, col = i & 511;
    int cc = col >> 7, jj = col & 127, nt = jj >> 4, nin = jj & 15;
    int kt = k >> 5, kin = k & 31;
    dst[(size_t)((cc * 16 + kt) * 8 + nt) * 512 + swz_tile(kin, nin)] =
        (__bf16)src[i];
  }
}

// --------------------------------------------------- kernel 1: flash stage 1
__global__ __launch_bounds__(256) void k_stage1(
    const float* __restrict__ x, const __bf16* __restrict__ Qb,
    const __bf16* __restrict__ Wkvb, const float* __restrict__ kv1_b,
    float* __restrict__ part) {
  extern __shared__ char smem[];
  __bf16* sQ   = (__bf16*)smem;          // 256x64 row-major (A operand)
  __bf16* sWk  = sQ + 256 * 64;          // swizzled 64 tiles
  __bf16* sWv  = sWk + 512 * 64;         // swizzled 64 tiles
  __bf16* sX   = sWv + 512 * 64;         // 64x512 row-major (A operand)
  __bf16* sK   = sX + 64 * 512;          // 64x64 row-major (K^T via frag_A)
  __bf16* sV   = sK + 64 * 64;           // 64x64 swizzled 8 tiles (B operand)
  float*  sS   = (float*)(sV + 64 * 64); // 256x64 scores -> probs
  float*  sMax = sS + 256 * 64;
  float*  sSum = sMax + 256;
  float*  sCorr = sSum + 256;

  const int tid = threadIdx.x, lane = tid & 31, wave = tid >> 5;
  const int idx = blockIdx.x;
  const int s = idx % SPLIT, h = (idx / SPLIT) % HEADS, b = idx / (SPLIT * HEADS);
  v8f vzero = {};

  {  // async DMA fills: weights pre-swizzled bf16 in ws, pure byte copies
    const __bf16* qg = Qb + (size_t)h * 256 * 64;
    const __bf16* kg = Wkvb + (size_t)h * 32768;
    const __bf16* vg = Wkvb + (size_t)(HEADS + h) * 32768;
    unsigned lq = lds_addr_of(sQ), lk = lds_addr_of(sWk), lv = lds_addr_of(sWv);
    for (int e = tid; e < 32768 / 8; e += 256) {
      async_cp16(lk + e * 16u, kg, e * 16u);
      async_cp16(lv + e * 16u, vg, e * 16u);
    }
    for (int e = tid; e < 256 * 64 / 8; e += 256) async_cp16(lq + e * 16u, qg, e * 16u);
    async_wait0();
  }
  sMax[tid] = -3.0e38f;
  sSum[tid] = 0.f;

  v8f acc[2][4];
#pragma unroll
  for (int i = 0; i < 2; ++i)
#pragma unroll
    for (int j = 0; j < 4; ++j) acc[i][j] = vzero;
  __syncthreads();

  for (int t = 0; t < CHUNK / 64; ++t) {
    const int row0 = s * CHUNK + t * 64;
    for (int e = tid; e < 64 * 512 / 4; e += 256) {  // f32->bf16 conversion fill
      int base = e * 4, r = base >> 9, c = base & 511;
      float4 v = *(const float4*)&x[((size_t)b * NN + row0 + r) * C_DIM + c];
      bf4 o;
      o[0] = (__bf16)v.x; o[1] = (__bf16)v.y; o[2] = (__bf16)v.z; o[3] = (__bf16)v.w;
      *(bf4*)&sX[base] = o;
    }
    __syncthreads();
    // K1/V1: waves 0-3 -> K tiles (mi=wave), waves 4-7 -> V tiles.
    // A fragment hoisted across the 4 N-tiles per k-step.
    {
      const int mat = wave >> 2, mi = wave & 3;
      const __bf16* W = mat ? sWv : sWk;
      v8f kv[4];
#pragma unroll
      for (int ni = 0; ni < 4; ++ni) kv[ni] = vzero;
#pragma unroll
      for (int kk = 0; kk < 16; ++kk) {
        v16bf af = frag_A(sX, 512, mi * 16, kk * 32, lane);
#pragma unroll
        for (int ni = 0; ni < 4; ++ni)
          kv[ni] = wmma_bf16(af, frag_Bs(W, kk * 4 + ni, lane), kv[ni]);
      }
      int rbase = mi * 16 + ((lane >> 4) << 3);
#pragma unroll
      for (int ni = 0; ni < 4; ++ni) {
        int col = ni * 16 + (lane & 15);
        float bv = kv1_b[(mat ? 512 : 0) + h * 64 + col];
        if (mat) {
#pragma unroll
          for (int r = 0; r < 8; ++r)
            stB_swz(sV, rbase + r, col, (__bf16)(kv[ni][r] + bv));
        } else {
#pragma unroll
          for (int r = 0; r < 8; ++r)
            sK[(rbase + r) * 64 + col] = (__bf16)(kv[ni][r] + bv);
        }
      }
    }
    __syncthreads();
    // scores(256x64) = Q @ K1^T * scale: wave owns m-tiles {2w,2w+1} x 4 n-tiles;
    // K^T B-fragment shared across the two m-tiles.
    {
      v8f sc[2][4];
#pragma unroll
      for (int i2 = 0; i2 < 2; ++i2)
#pragma unroll
        for (int ni = 0; ni < 4; ++ni) sc[i2][ni] = vzero;
#pragma unroll
      for (int kk = 0; kk < 2; ++kk) {
        v16bf aq0 = frag_A(sQ, 64, (wave * 2) * 16, kk * 32, lane);
        v16bf aq1 = frag_A(sQ, 64, (wave * 2 + 1) * 16, kk * 32, lane);
#pragma unroll
        for (int ni = 0; ni < 4; ++ni) {
          v16bf bk = frag_A(sK, 64, ni * 16, kk * 32, lane);
          sc[0][ni] = wmma_bf16(aq0, bk, sc[0][ni]);
          sc[1][ni] = wmma_bf16(aq1, bk, sc[1][ni]);
        }
      }
#pragma unroll
      for (int i2 = 0; i2 < 2; ++i2) {
        int rbase = (wave * 2 + i2) * 16 + ((lane >> 4) << 3);
#pragma unroll
        for (int ni = 0; ni < 4; ++ni) {
          int col = ni * 16 + (lane & 15);
#pragma unroll
          for (int r = 0; r < 8; ++r)
            sS[(rbase + r) * 64 + col] = sc[i2][ni][r] * 0.125f;
        }
      }
    }
    __syncthreads();
    // online softmax: one thread per m row
    {
      float* row = sS + tid * 64;
      float rm = -3.0e38f;
      for (int n = 0; n < 64; ++n) rm = fmaxf(rm, row[n]);
      float om = sMax[tid], nm = fmaxf(om, rm);
      float corr = __expf(om - nm), sum = 0.f;
      for (int n = 0; n < 64; ++n) {
        float p = __expf(row[n] - nm);
        row[n] = p;
        sum += p;
      }
      sMax[tid] = nm;
      sSum[tid] = sSum[tid] * corr + sum;
      sCorr[tid] = corr;
    }
    __syncthreads();
    // rescale accumulators
#pragma unroll
    for (int i = 0; i < 2; ++i) {
      int rbase = (wave * 2 + i) * 16 + ((lane >> 4) << 3);
      float cr[8];
#pragma unroll
      for (int r = 0; r < 8; ++r) cr[r] = sCorr[rbase + r];
#pragma unroll
      for (int j = 0; j < 4; ++j)
#pragma unroll
        for (int r = 0; r < 8; ++r) acc[i][j][r] *= cr[r];
    }
    // out1 += P @ V1: A (probs) hoisted per m-tile, B shared across m-tiles
#pragma unroll
    for (int kk = 0; kk < 2; ++kk) {
      v16bf a0 = frag_A(sS, 64, (wave * 2) * 16, kk * 32, lane);
      v16bf a1 = frag_A(sS, 64, (wave * 2 + 1) * 16, kk * 32, lane);
#pragma unroll
      for (int j = 0; j < 4; ++j) {
        v16bf bvf = frag_Bs(sV, kk * 4 + j, lane);
        acc[0][j] = wmma_bf16(a0, bvf, acc[0][j]);
        acc[1][j] = wmma_bf16(a1, bvf, acc[1][j]);
      }
    }
    __syncthreads();
  }
  float* pa = part + (size_t)((b * HEADS + h) * SPLIT + s) * PART_STRIDE;
#pragma unroll
  for (int i = 0; i < 2; ++i) {
    int rbase = (wave * 2 + i) * 16 + ((lane >> 4) << 3);
#pragma unroll
    for (int j = 0; j < 4; ++j) {
      int col = j * 16 + (lane & 15);
#pragma unroll
      for (int r = 0; r < 8; ++r) pa[(rbase + r) * 64 + col] = acc[i][j][r];
    }
  }
  pa[N_TOK * DH + tid] = sMax[tid];
  pa[N_TOK * DH + 256 + tid] = sSum[tid];
}

// --------------------------------------- kernel 2: merge splits + kv2 -> K2,V2
__global__ __launch_bounds__(256) void k_combine_kv2(
    const float* __restrict__ part, const float* __restrict__ kv2_w,
    const float* __restrict__ kv2_b, __bf16* __restrict__ k2b,
    __bf16* __restrict__ v2b) {
  extern __shared__ char smem[];
  float* sO = (float*)smem;      // 256x64 out1
  float* sW = sO + 256 * 64;     // 64x128 kv2_w
  const int tid = threadIdx.x, bh = blockIdx.x;
  const float* base = part + (size_t)bh * SPLIT * PART_STRIDE;
  for (int e = tid; e < 64 * 128; e += 256) sW[e] = kv2_w[e];
  const int m = tid;
  float w[SPLIT], M = -3.0e38f;
  for (int s2 = 0; s2 < SPLIT; ++s2)
    M = fmaxf(M, base[s2 * PART_STRIDE + N_TOK * DH + m]);
  float L = 0.f;
  for (int s2 = 0; s2 < SPLIT; ++s2) {
    float e = __expf(base[s2 * PART_STRIDE + N_TOK * DH + m] - M);
    w[s2] = e;
    L += e * base[s2 * PART_STRIDE + N_TOK * DH + 256 + m];
  }
  float invL = 1.f / L;
  for (int c = 0; c < 64; ++c) {
    float o = 0.f;
    for (int s2 = 0; s2 < SPLIT; ++s2)
      o += w[s2] * base[s2 * PART_STRIDE + m * 64 + c];
    sO[m * 64 + c] = o * invL;
  }
  __syncthreads();
  for (int j = 0; j < 128; ++j) {
    float a = kv2_b[j];
    for (int c = 0; c < 64; ++c) a += sO[m * 64 + c] * sW[c * 128 + j];
    if (j < 64)
      k2b[(size_t)bh * N_TOK * DH + m * 64 + j] = (__bf16)a;   // row-major
    else
      stB_swz(v2b + (size_t)bh * N_TOK * DH, m, j - 64, (__bf16)a);  // swizzled
  }
}

// ------------------------- kernel 3: q3 proj + stage-2 attention (fused/head)
__global__ __launch_bounds__(256) void k_stage2(
    const float* __restrict__ x, const __bf16* __restrict__ q3wb,
    const float* __restrict__ q3_b, const __bf16* __restrict__ k2b,
    const __bf16* __restrict__ v2b, float* __restrict__ ctx) {
  extern __shared__ char smem[];
  __bf16* sX  = (__bf16*)smem;                 // 64x512 row-major
  char*   uni = smem + 64 * 512 * 2;           // union: sW swz / sS f32[64x256]
  __bf16* sW  = (__bf16*)uni;
  float*  sS  = (float*)uni;
  __bf16* sQ3 = (__bf16*)(uni + 64 * 256 * 4); // 64x64 row-major
  __bf16* sK2 = sQ3 + 64 * 64;                 // 256x64 row-major
  __bf16* sV2 = sK2 + 256 * 64;                // swizzled 32 tiles
  const int tid = threadIdx.x, lane = tid & 31, wave = tid >> 5;
  const int tile = blockIdx.x;
  const int b = tile / (NN / 64), r0 = (tile % (NN / 64)) * 64;
  v8f vzero = {};
  for (int e = tid; e < 64 * 512 / 4; e += 256) {  // f32->bf16 conversion fill
    int base = e * 4, r = base >> 9, c = base & 511;
    float4 v = *(const float4*)&x[((size_t)b * NN + r0 + r) * C_DIM + c];
    bf4 o;
    o[0] = (__bf16)v.x; o[1] = (__bf16)v.y; o[2] = (__bf16)v.z; o[3] = (__bf16)v.w;
    *(bf4*)&sX[base] = o;
  }
  for (int h = 0; h < HEADS; ++h) {
    __syncthreads();
    {  // async DMA fills (8x per WG)
      const __bf16* wg = q3wb + (size_t)h * 32768;
      const __bf16* kg = k2b + (size_t)(b * HEADS + h) * 16384;
      const __bf16* vg = v2b + (size_t)(b * HEADS + h) * 16384;
      unsigned lw = lds_addr_of(sW), lk = lds_addr_of(sK2), lv = lds_addr_of(sV2);
      for (int e = tid; e < 32768 / 8; e += 256) async_cp16(lw + e * 16u, wg, e * 16u);
      for (int e = tid; e < 16384 / 8; e += 256) {
        async_cp16(lk + e * 16u, kg, e * 16u);
        async_cp16(lv + e * 16u, vg, e * 16u);
      }
      async_wait0();
    }
    __syncthreads();
    // q3(64x64) = sX @ sW + bias: wave -> (mi=wave>>1, ni0=(wave&1)*2), A hoisted
    {
      const int mi = wave >> 1, ni0 = (wave & 1) * 2;
      v8f q[2] = {vzero, vzero};
#pragma unroll
      for (int kk = 0; kk < 16; ++kk) {
        v16bf af = frag_A(sX, 512, mi * 16, kk * 32, lane);
#pragma unroll
        for (int u = 0; u < 2; ++u)
          q[u] = wmma_bf16(af, frag_Bs(sW, kk * 4 + ni0 + u, lane), q[u]);
      }
      int rbase = mi * 16 + ((lane >> 4) << 3);
#pragma unroll
      for (int u = 0; u < 2; ++u) {
        int col = (ni0 + u) * 16 + (lane & 15);
        float bvq = q3_b[h * 64 + col];
#pragma unroll
        for (int r = 0; r < 8; ++r)
          sQ3[(rbase + r) * 64 + col] = (__bf16)(q[u][r] + bvq);
      }
    }
    __syncthreads();  // sW dead -> sS may overwrite union
    // scores(64x256) = q3 @ K2^T * scale: wave -> (mi=wave>>1, ni0=(wave&1)*8)
    {
      const int mi = wave >> 1, ni0 = (wave & 1) * 8;
      v8f sc[8];
#pragma unroll
      for (int u = 0; u < 8; ++u) sc[u] = vzero;
#pragma unroll
      for (int kk = 0; kk < 2; ++kk) {
        v16bf aq = frag_A(sQ3, 64, mi * 16, kk * 32, lane);
#pragma unroll
        for (int u = 0; u < 8; ++u)
          sc[u] = wmma_bf16(aq, frag_A(sK2, 64, (ni0 + u) * 16, kk * 32, lane),
                            sc[u]);
      }
      int rbase = mi * 16 + ((lane >> 4) << 3);
#pragma unroll
      for (int u = 0; u < 8; ++u) {
        int col = (ni0 + u) * 16 + (lane & 15);
#pragma unroll
        for (int r = 0; r < 8; ++r)
          sS[(rbase + r) * 256 + col] = sc[u][r] * 0.125f;
      }
    }
    __syncthreads();
    if (tid < 64) {  // full softmax over m=256
      float* row = sS + tid * 256;
      float rm = -3.0e38f;
      for (int n2 = 0; n2 < 256; ++n2) rm = fmaxf(rm, row[n2]);
      float sum = 0.f;
      for (int n2 = 0; n2 < 256; ++n2) {
        float p = __expf(row[n2] - rm);
        row[n2] = p;
        sum += p;
      }
      float inv = 1.f / sum;
      for (int n2 = 0; n2 < 256; ++n2) row[n2] *= inv;
    }
    __syncthreads();
    // out_h(64x64) = P @ V2: wave -> (mi=wave>>1, ni0=(wave&1)*2), A hoisted
    {
      const int mi = wave >> 1, ni0 = (wave & 1) * 2;
      v8f o[2] = {vzero, vzero};
#pragma unroll
      for (int kk = 0; kk < 8; ++kk) {
        v16bf ap = frag_A(sS, 256, mi * 16, kk * 32, lane);
#pragma unroll
        for (int u = 0; u < 2; ++u)
          o[u] = wmma_bf16(ap, frag_Bs(sV2, kk * 4 + ni0 + u, lane), o[u]);
      }
      int rbase = mi * 16 + ((lane >> 4) << 3);
#pragma unroll
      for (int u = 0; u < 2; ++u) {
        int col = (ni0 + u) * 16 + (lane & 15);
#pragma unroll
        for (int r = 0; r < 8; ++r)
          ctx[((size_t)b * NN + r0 + rbase + r) * C_DIM + h * 64 + col] = o[u][r];
      }
    }
  }
}

// ------------------------------- kernel 4: final projection, in-place on d_out
__global__ __launch_bounds__(256) void k_outproj(
    float* __restrict__ out, const __bf16* __restrict__ outwb,
    const float* __restrict__ out_b) {
  extern __shared__ char smem[];
  __bf16* sC = (__bf16*)smem;     // 64x512 row-major ctx tile
  __bf16* sW = sC + 64 * 512;     // swizzled 128 tiles per chunk
  const int tid = threadIdx.x, lane = tid & 31, wave = tid >> 5;
  const size_t r0 = (size_t)blockIdx.x * 64;
  v8f vzero = {};
  for (int e = tid; e < 64 * 512 / 4; e += 256) {  // f32->bf16 conversion fill
    int base = e * 4, r = base >> 9, c = base & 511;
    float4 v = *(const float4*)&out[(r0 + r) * C_DIM + c];
    bf4 o;
    o[0] = (__bf16)v.x; o[1] = (__bf16)v.y; o[2] = (__bf16)v.z; o[3] = (__bf16)v.w;
    *(bf4*)&sC[base] = o;
  }
  for (int cc = 0; cc < 4; ++cc) {
    __syncthreads();
    {  // async DMA fill of weight chunk
      const __bf16* wg = outwb + (size_t)cc * 65536;
      unsigned lw = lds_addr_of(sW);
      for (int e = tid; e < 65536 / 8; e += 256) async_cp16(lw + e * 16u, wg, e * 16u);
      async_wait0();
    }
    __syncthreads();
    // wave -> (mi=wave>>1, ni0=(wave&1)*4), A hoisted across 4 n-tiles
    {
      const int mi = wave >> 1, ni0 = (wave & 1) * 4;
      v8f o[4];
#pragma unroll
      for (int u = 0; u < 4; ++u) o[u] = vzero;
#pragma unroll
      for (int kk = 0; kk < 16; ++kk) {
        v16bf af = frag_A(sC, 512, mi * 16, kk * 32, lane);
#pragma unroll
        for (int u = 0; u < 4; ++u)
          o[u] = wmma_bf16(af, frag_Bs(sW, kk * 8 + ni0 + u, lane), o[u]);
      }
      int rbase = mi * 16 + ((lane >> 4) << 3);
#pragma unroll
      for (int u = 0; u < 4; ++u) {
        int col = cc * 128 + (ni0 + u) * 16 + (lane & 15);
        float bv = out_b[col];
#pragma unroll
        for (int r = 0; r < 8; ++r)
          out[(r0 + rbase + r) * C_DIM + col] = o[u][r] + bv;
      }
    }
  }
}

// -------------------------------------------------------------------- launch
extern "C" void kernel_launch(void* const* d_in, const int* in_sizes, int n_in,
                              void* d_out, int out_size, void* d_ws, size_t ws_size,
                              hipStream_t stream) {
  (void)in_sizes; (void)n_in; (void)out_size; (void)ws_size;
  const float* x     = (const float*)d_in[0];
  const float* Q     = (const float*)d_in[1];
  const float* kv1_w = (const float*)d_in[2];
  const float* kv1_b = (const float*)d_in[3];
  const float* kv2_w = (const float*)d_in[4];
  const float* kv2_b = (const float*)d_in[5];
  const float* q3_w  = (const float*)d_in[6];
  const float* q3_b  = (const float*)d_in[7];
  const float* out_w = (const float*)d_in[8];
  const float* out_b = (const float*)d_in[9];
  float* out = (float*)d_out;

  char* ws = (char*)d_ws;
  __bf16* Qb    = (__bf16*)(ws + 0);                 // linear bf16
  __bf16* Wkvb  = (__bf16*)(ws + 262144);            // swizzled per (mat,head)
  __bf16* q3wb  = (__bf16*)(ws + 1310720);           // swizzled per head
  __bf16* outwb = (__bf16*)(ws + 1835008);           // swizzled per col-chunk
  float*  part  = (float*) (ws + 2359296);
  __bf16* k2b   = (__bf16*)(ws + 19660800);          // row-major
  __bf16* v2b   = (__bf16*)(ws + 20709376);          // swizzled per (b,h)

  k_cvt_lin<<<256, 256, 0, stream>>>(Q, Qb, HEADS * N_TOK * DH);
  k_cvt_kv <<<512, 256, 0, stream>>>(kv1_w, Wkvb);
  k_cvt_q3 <<<256, 256, 0, stream>>>(q3_w, q3wb);
  k_cvt_out<<<256, 256, 0, stream>>>(out_w, outwb);

  size_t lds1 = (256*64 + 512*64*2 + 64*512 + 64*64*2) * 2 + 256*64*4 + 3*256*4;
  k_stage1<<<BB * HEADS * SPLIT, 256, lds1, stream>>>(x, Qb, Wkvb, kv1_b, part);

  size_t lds2 = 256*64*4 + 64*128*4;
  k_combine_kv2<<<BB * HEADS, 256, lds2, stream>>>(part, kv2_w, kv2_b, k2b, v2b);

  size_t lds3 = 64*512*2 + 64*256*4 + 64*64*2 + 2 * 256*64*2;
  k_stage2<<<BB * (NN / 64), 256, lds3, stream>>>(x, q3wb, q3_b, k2b, v2b, out);

  size_t lds4 = 64*512*2 + 512*128*2;
  k_outproj<<<(BB * NN) / 64, 256, lds4, stream>>>(out, outwb, out_b);
}
